// PKGNNEncoder_17617955848712
// MI455X (gfx1250) — matrix operand order repaired and checked by reading, not compiled, not run on hardware
//
#include <hip/hip_runtime.h>
#include <hip/hip_bf16.h>
#include <math.h>

// ---------------------------------------------------------------------------
// PKGNNEncoder for MI455X (gfx1250, wave32)
//  - GEMMs via v_wmma_f32_16x16x32_f16 (f32 accumulate), K templated so the
//    k-chunk loop fully unrolls (4 WMMAs for K=128, 2 for K=64)
//  - edge aggregation via global_atomic_add_f32 (xw/agg are 25.6MB each ->
//    fully L2-resident on the 192MB L2; atomics resolve at L2)
//  - wave32 shuffle reductions for LayerNorm / MLP head
// ---------------------------------------------------------------------------

typedef __attribute__((ext_vector_type(16))) _Float16 v16h;
typedef __attribute__((ext_vector_type(8)))  float    v8f;

#define TPB 256

static __host__ __device__ inline int cdiv_i(int a, int b) { return (a + b - 1) / b; }

// ---- degree / normalization (layer-invariant, computed once) --------------

__global__ void init_deg_kernel(float* __restrict__ deg, int N) {
    int i = blockIdx.x * blockDim.x + threadIdx.x;
    if (i < N) deg[i] = 1.0f;                    // self-loop weight 1.0
}

__global__ void deg_accum_kernel(float* __restrict__ deg,
                                 const long long* __restrict__ dst,
                                 const float* __restrict__ ew, int E) {
    int i = blockIdx.x * blockDim.x + threadIdx.x;
    if (i < E) atomicAdd(&deg[(int)dst[i]], ew[i]);
}

__global__ void dis_kernel(float* __restrict__ deg, int N) {
    int i = blockIdx.x * blockDim.x + threadIdx.x;
    if (i < N) {
        float d = deg[i];
        deg[i] = d > 0.0f ? rsqrtf(d) : 0.0f;    // in-place: deg -> dis
    }
}

__global__ void norm_kernel(float* __restrict__ norm,
                            const float* __restrict__ dis,
                            const long long* __restrict__ src,
                            const long long* __restrict__ dst,
                            const float* __restrict__ ew, int E) {
    int i = blockIdx.x * blockDim.x + threadIdx.x;
    if (i < E) norm[i] = dis[(int)src[i]] * ew[i] * dis[(int)dst[i]];
}

// ---- WMMA GEMM:  Xw[N,64] = H[N,K] @ W[K,64]  -----------------------------
// One wave per 16x16 output tile; K consumed 32 at a time (fully unrolled).
// A layout (16x32 f16): lane L (m=L%16, hi=L/16): halves 0..7  <- K=kb+hi*8+j
//                                                 halves 8..15 <- K=kb+16+hi*8+j
// B layout (32x16 f16): lane L (n=L%16, hi=L/16): same K mapping as A
// C/D (16x16 f32): lane L, vgpr r -> (m = r + 8*hi, n = L%16)

template <int K>
__global__ __launch_bounds__(TPB) void gemm_wmma_kernel(
        const float* __restrict__ Hm, const float* __restrict__ W,
        float* __restrict__ Xw, int N) {
    int wave = (blockIdx.x * blockDim.x + threadIdx.x) >> 5;
    int lane = threadIdx.x & 31;
    int mtiles = (N + 15) >> 4;
    if (wave >= mtiles * 4) return;              // wave-uniform: EXEC stays all-1s
    int mtile = wave >> 2;
    int ntile = wave & 3;
    int hi = lane >> 4;
    int ml = lane & 15;
    int row = mtile * 16 + ml;
    int rc  = row < N ? row : N - 1;             // clamp OOB rows (stores guarded)
    const float* arow = Hm + (size_t)rc * K;
    int ncol = ntile * 16 + ml;

    v8f c = {};
#pragma unroll
    for (int kb = 0; kb < K; kb += 32) {
        int kA = kb + hi * 8;
        float4 a0 = *reinterpret_cast<const float4*>(arow + kA);
        float4 a1 = *reinterpret_cast<const float4*>(arow + kA + 4);
        float4 a2 = *reinterpret_cast<const float4*>(arow + kA + 16);
        float4 a3 = *reinterpret_cast<const float4*>(arow + kA + 20);
        v16h a;
        a[0]=(_Float16)a0.x;  a[1]=(_Float16)a0.y;  a[2]=(_Float16)a0.z;  a[3]=(_Float16)a0.w;
        a[4]=(_Float16)a1.x;  a[5]=(_Float16)a1.y;  a[6]=(_Float16)a1.z;  a[7]=(_Float16)a1.w;
        a[8]=(_Float16)a2.x;  a[9]=(_Float16)a2.y;  a[10]=(_Float16)a2.z; a[11]=(_Float16)a2.w;
        a[12]=(_Float16)a3.x; a[13]=(_Float16)a3.y; a[14]=(_Float16)a3.z; a[15]=(_Float16)a3.w;

        int kB = kb + hi * 8;
        v16h b;
#pragma unroll
        for (int j = 0; j < 16; ++j)
            b[j] = (_Float16)W[(size_t)(kB + j + (j >= 8 ? 8 : 0)) * 64 + ncol];

        // 8 args: (neg_a, A, neg_b, B, c_mod, C, reuse_a, reuse_b)
        c = __builtin_amdgcn_wmma_f32_16x16x32_f16(
                false, a, false, b, (short)0, c, false, false);
    }

    // Store: wave-uniform full-tile fast path (N%16==0 -> always taken),
    // avoids per-row exec-mask churn.
    if (mtile * 16 + 16 <= N) {
        float* outp = Xw + (size_t)(mtile * 16 + 8 * hi) * 64 + ncol;
#pragma unroll
        for (int r = 0; r < 8; ++r)
            outp[(size_t)r * 64] = c[r];
    } else {
#pragma unroll
        for (int r = 0; r < 8; ++r) {
            int m = mtile * 16 + r + 8 * hi;
            if (m < N) Xw[(size_t)m * 64 + ncol] = c[r];
        }
    }
}

// ---- aggregation -----------------------------------------------------------

// agg[n,f] = xw[n,f] * dis[n]^2    (self-loop term, also initializes agg)
__global__ void agg_init_kernel(float* __restrict__ agg,
                                const float* __restrict__ xw,
                                const float* __restrict__ dis, int total) {
    int i = blockIdx.x * blockDim.x + threadIdx.x;
    if (i < total) {
        float di = dis[i >> 6];
        agg[i] = xw[i] * di * di;
    }
}

// one wave per edge: agg[dst] += xw[src] * norm[e]  (2 floats per lane)
__global__ void scatter_kernel(float* __restrict__ agg,
                               const float* __restrict__ xw,
                               const long long* __restrict__ src,
                               const long long* __restrict__ dst,
                               const float* __restrict__ norm, int E) {
    int wave = (blockIdx.x * blockDim.x + threadIdx.x) >> 5;
    int lane = threadIdx.x & 31;
    if (wave >= E) return;
    int s = (int)src[wave];
    int d = (int)dst[wave];
    float w = norm[wave];
    float2 v = *reinterpret_cast<const float2*>(xw + (size_t)s * 64 + lane * 2);
    float* ad = agg + (size_t)d * 64 + lane * 2;
    atomicAdd(ad,     v.x * w);
    atomicAdd(ad + 1, v.y * w);
}

// ---- bias + LayerNorm + ReLU (+ residual) ---------------------------------
// one wave per node, 2 features per lane (f, f+32), wave32 shuffle reductions

__global__ void ln_relu_kernel(float* __restrict__ out,
                               const float* __restrict__ agg,
                               const float* __restrict__ bias,
                               const float* __restrict__ g,
                               const float* __restrict__ be,
                               const float* __restrict__ resid, int N) {
    int wave = (blockIdx.x * blockDim.x + threadIdx.x) >> 5;
    int lane = threadIdx.x & 31;
    if (wave >= N) return;
    const float* a = agg + (size_t)wave * 64;
    int f0 = lane, f1 = lane + 32;
    float t0 = a[f0] + bias[f0];
    float t1 = a[f1] + bias[f1];

    float s = t0 + t1;
#pragma unroll
    for (int m = 16; m >= 1; m >>= 1) s += __shfl_xor(s, m, 32);
    float mean = s * (1.0f / 64.0f);
    float d0 = t0 - mean, d1 = t1 - mean;
    float v = d0 * d0 + d1 * d1;
#pragma unroll
    for (int m = 16; m >= 1; m >>= 1) v += __shfl_xor(v, m, 32);
    float inv = rsqrtf(v * (1.0f / 64.0f) + 1e-5f);

    float y0 = fmaxf(d0 * inv * g[f0] + be[f0], 0.0f);
    float y1 = fmaxf(d1 * inv * g[f1] + be[f1], 0.0f);
    if (resid) {
        y0 += resid[(size_t)wave * 64 + f0];
        y1 += resid[(size_t)wave * 64 + f1];
    }
    out[(size_t)wave * 64 + f0] = y0;
    out[(size_t)wave * 64 + f1] = y1;
}

// ---- pk predictor: relu(emb @ Wp1 + bp1) @ Wp2 + bp2 ----------------------
// one wave per node; lane j owns hidden unit j (H/2 == 32 == wave32)

__global__ void predictor_kernel(float* __restrict__ pk,
                                 const float* __restrict__ emb,
                                 const float* __restrict__ Wp1,
                                 const float* __restrict__ bp1,
                                 const float* __restrict__ Wp2,
                                 const float* __restrict__ bp2, int N) {
    int wave = (blockIdx.x * blockDim.x + threadIdx.x) >> 5;
    int lane = threadIdx.x & 31;
    if (wave >= N) return;
    const float* e = emb + (size_t)wave * 64;
    float h = bp1[lane];
#pragma unroll
    for (int k = 0; k < 64; ++k)
        h = fmaf(e[k], Wp1[k * 32 + lane], h);
    h = fmaxf(h, 0.0f);
    float p = h * Wp2[lane];
#pragma unroll
    for (int m = 16; m >= 1; m >>= 1) p += __shfl_xor(p, m, 32);
    if (lane == 0) pk[wave] = p + bp2[0];
}

// ---------------------------------------------------------------------------

extern "C" void kernel_launch(void* const* d_in, const int* in_sizes, int n_in,
                              void* d_out, int out_size, void* d_ws, size_t ws_size,
                              hipStream_t stream) {
    const int D = 128, H = 64;
    const int N = in_sizes[0] / D;
    const int E = in_sizes[2];

    const float*     x   = (const float*)d_in[0];
    const long long* ei  = (const long long*)d_in[1];   // int64 [2,E]
    const float*     ew  = (const float*)d_in[2];
    const float*     W1  = (const float*)d_in[3];
    const float*     b1  = (const float*)d_in[4];
    const float*     W2  = (const float*)d_in[5];
    const float*     b2  = (const float*)d_in[6];
    const float*     W3  = (const float*)d_in[7];
    const float*     b3  = (const float*)d_in[8];
    const float*     g1  = (const float*)d_in[9];
    const float*     be1 = (const float*)d_in[10];
    const float*     g2  = (const float*)d_in[11];
    const float*     be2 = (const float*)d_in[12];
    const float*     g3  = (const float*)d_in[13];
    const float*     be3 = (const float*)d_in[14];
    const float*     Wp1 = (const float*)d_in[15];
    const float*     bp1 = (const float*)d_in[16];
    const float*     Wp2 = (const float*)d_in[17];
    const float*     bp2 = (const float*)d_in[18];

    const long long* srcI = ei;          // edge_index[0]
    const long long* dstI = ei + E;      // edge_index[1]

    // workspace layout
    float* dis  = (float*)d_ws;                  // N   (deg -> dis, in place)
    float* nrm  = dis + N;                       // E
    float* xw   = nrm + E;                       // N*H
    float* agg  = xw  + (size_t)N * H;           // N*H
    float* hA   = agg + (size_t)N * H;           // N*H
    float* hB   = hA  + (size_t)N * H;           // N*H

    float* emb_out = (float*)d_out;              // N*H
    float* pk_out  = emb_out + (size_t)N * H;    // N

    const int gN   = cdiv_i(N, TPB);
    const int gE   = cdiv_i(E, TPB);
    const int gNH  = cdiv_i(N * H, TPB);
    const int gWN  = cdiv_i(N, TPB / 32);        // wave per node
    const int gWE  = cdiv_i(E, TPB / 32);        // wave per edge
    const int tiles = cdiv_i(N, 16) * 4;
    const int gT   = cdiv_i(tiles, TPB / 32);    // wave per WMMA tile

    // ---- symmetric normalization (once; reused by all layers) ----
    init_deg_kernel<<<gN, TPB, 0, stream>>>(dis, N);
    deg_accum_kernel<<<gE, TPB, 0, stream>>>(dis, dstI, ew, E);
    dis_kernel<<<gN, TPB, 0, stream>>>(dis, N);
    norm_kernel<<<gE, TPB, 0, stream>>>(nrm, dis, srcI, dstI, ew, E);

    // ---- layer 1: x[N,128] -> hA[N,64] ----
    gemm_wmma_kernel<128><<<gT, TPB, 0, stream>>>(x, W1, xw, N);
    agg_init_kernel<<<gNH, TPB, 0, stream>>>(agg, xw, dis, N * H);
    scatter_kernel<<<gWE, TPB, 0, stream>>>(agg, xw, srcI, dstI, nrm, E);
    ln_relu_kernel<<<gWN, TPB, 0, stream>>>(hA, agg, b1, g1, be1, nullptr, N);

    // ---- layer 2: hA -> hB (residual) ----
    gemm_wmma_kernel<64><<<gT, TPB, 0, stream>>>(hA, W2, xw, N);
    agg_init_kernel<<<gNH, TPB, 0, stream>>>(agg, xw, dis, N * H);
    scatter_kernel<<<gWE, TPB, 0, stream>>>(agg, xw, srcI, dstI, nrm, E);
    ln_relu_kernel<<<gWN, TPB, 0, stream>>>(hB, agg, b2, g2, be2, hA, N);

    // ---- layer 3: hB -> emb_out (residual), straight into d_out ----
    gemm_wmma_kernel<64><<<gT, TPB, 0, stream>>>(hB, W3, xw, N);
    agg_init_kernel<<<gNH, TPB, 0, stream>>>(agg, xw, dis, N * H);
    scatter_kernel<<<gWE, TPB, 0, stream>>>(agg, xw, srcI, dstI, nrm, E);
    ln_relu_kernel<<<gWN, TPB, 0, stream>>>(emb_out, agg, b3, g3, be3, hB, N);

    // ---- pk head ----
    predictor_kernel<<<gWN, TPB, 0, stream>>>(pk_out, emb_out, Wp1, bp1, Wp2, bp2, N);
}
